// EncoderLayer_54142357733642
// MI455X (gfx1250) — matrix operand order repaired and checked by reading, not compiled, not run on hardware
//
#include <hip/hip_runtime.h>

typedef _Float16 f16;
typedef __attribute__((ext_vector_type(16))) _Float16 v16h;
typedef __attribute__((ext_vector_type(8)))  float    v8f;
typedef __attribute__((ext_vector_type(4)))  unsigned int u32x4;
typedef __attribute__((ext_vector_type(4)))  int v4i;

#define BB 4
#define SS 2048
#define DD 1024
#define HH 16
#define DKK 64
#define FFD 4096

#if defined(__has_builtin)
#if __has_builtin(__builtin_amdgcn_global_load_async_to_lds_b128)
#define HAS_ASYNC_LDS 1
#endif
#if __has_builtin(__builtin_amdgcn_s_wait_asynccnt)
#define HAS_WAIT_ASYNC 1
#endif
#endif

union AF { v16h v; u32x4 q[2]; };

// A-operand fragment (16x32 f16): lane half selects K base {0,8}; two 16B chunks at +0 and +16 elems.
__device__ inline v16h load_frag_a(const f16* p, int half) {
  AF f; const int kb = half * 8;
  f.q[0] = *(const u32x4*)(p + kb);
  f.q[1] = *(const u32x4*)(p + kb + 16);
  return f.v;
}
// B-operand fragment (32x16 f16), stored row=N, K contiguous: lane half selects K base {0,8}; two 16B chunks at +0 and +16 elems.
__device__ inline v16h load_frag_b(const f16* p, int half) {
  AF f; const int kb = half * 8;
  f.q[0] = *(const u32x4*)(p + kb);
  f.q[1] = *(const u32x4*)(p + kb + 16);
  return f.v;
}
__device__ inline v8f wmma16(v16h a, v16h b, v8f c) {
  return __builtin_amdgcn_wmma_f32_16x16x32_f16(false, a, false, b, (short)0, c, false, false);
}

// 16-byte global -> LDS copy: CDNA5 async DMA path when available (ASYNCcnt),
// else register relay (tracked by LOADcnt/DScnt, synced by barrier).
// Builtin signature (from probe diagnostic): (v4i addrspace(1)*, v4i addrspace(3)*, imm, imm).
__device__ inline void cp16_async(const f16* g, f16* l) {
#if defined(HAS_ASYNC_LDS)
  __builtin_amdgcn_global_load_async_to_lds_b128(
      (__attribute__((address_space(1))) v4i*)((v4i*)(void*)g),
      (__attribute__((address_space(3))) v4i*)((v4i*)(void*)l), 0, 0);
#else
  *(u32x4*)l = *(const u32x4*)g;
#endif
}
__device__ inline void wait_async_lds() {
#if defined(HAS_WAIT_ASYNC)
  __builtin_amdgcn_s_wait_asynccnt(0);
#elif defined(HAS_ASYNC_LDS)
  asm volatile("s_wait_asynccnt 0x0" ::: "memory");
#endif
}

// ---------------------------------------------------------------------------
// Tiled WMMA GEMM:  C[M,N] = A[M,K](f16) * Bt[N,K](f16)^T + bias, opt ReLU.
// 256 threads = 8 waves; block tile 128x128; K-step 32; wave tile 32x64.
// Double-buffered LDS, async global->LDS copies issued one stage ahead.
// ---------------------------------------------------------------------------
template<bool OUT16, bool RELU>
__global__ __launch_bounds__(256)
void gemm_kernel(const f16* __restrict__ A, const f16* __restrict__ Bt,
                 const float* __restrict__ bias, void* __restrict__ Cout,
                 int M, int N, int K)
{
  __shared__ f16 As[2][128 * 32];
  __shared__ f16 Bs[2][128 * 32];
  const int t    = threadIdx.x;
  const int lane = t & 31, w = t >> 5;
  const int half = lane >> 4, r16 = lane & 15;
  const int wm   = w >> 1, wn = w & 1;
  const int m0   = blockIdx.y * 128, n0 = blockIdx.x * 128;

  const int row = (t >> 2);            // this thread's tile row for staging
  const int kc  = (t & 3) * 8;         // this thread's 8-elem K chunk
  const int row2 = ((t + 256) >> 2), kc2 = ((t + 256) & 3) * 8;

  v8f acc[2][4] = {};

  // stage k-tile k0 into buffer `buf`
  auto stage = [&](int buf, int k0) {
    cp16_async(&A [(size_t)(m0 + row ) * K + k0 + kc ], &As[buf][row  * 32 + kc ]);
    cp16_async(&Bt[(size_t)(n0 + row ) * K + k0 + kc ], &Bs[buf][row  * 32 + kc ]);
    cp16_async(&A [(size_t)(m0 + row2) * K + k0 + kc2], &As[buf][row2 * 32 + kc2]);
    cp16_async(&Bt[(size_t)(n0 + row2) * K + k0 + kc2], &Bs[buf][row2 * 32 + kc2]);
  };

  stage(0, 0);
  wait_async_lds();
  __syncthreads();

  int buf = 0;
  for (int k0 = 0; k0 < K; k0 += 32) {
    if (k0 + 32 < K) stage(buf ^ 1, k0 + 32);   // prefetch next stage (async)

    v16h af[2], bf[4];
    #pragma unroll
    for (int mf = 0; mf < 2; ++mf)
      af[mf] = load_frag_a(&As[buf][(wm * 32 + mf * 16 + r16) * 32], half);
    #pragma unroll
    for (int nf = 0; nf < 4; ++nf)
      bf[nf] = load_frag_b(&Bs[buf][(wn * 64 + nf * 16 + r16) * 32], half);
    #pragma unroll
    for (int mf = 0; mf < 2; ++mf)
      #pragma unroll
      for (int nf = 0; nf < 4; ++nf)
        acc[mf][nf] = wmma16(af[mf], bf[nf], acc[mf][nf]);

    wait_async_lds();
    __syncthreads();
    buf ^= 1;
  }

  #pragma unroll
  for (int mf = 0; mf < 2; ++mf)
    #pragma unroll
    for (int nf = 0; nf < 4; ++nf) {
      const int n = n0 + wn * 64 + nf * 16 + r16;
      const float bv = bias ? bias[n] : 0.f;
      #pragma unroll
      for (int r = 0; r < 8; ++r) {
        const int m = m0 + wm * 32 + mf * 16 + half * 8 + r;
        float v = acc[mf][nf][r] + bv;
        if (RELU) v = fmaxf(v, 0.f);
        if (OUT16) ((f16*)Cout)[(size_t)m * N + n] = (f16)v;
        else       ((float*)Cout)[(size_t)m * N + n] = v;
      }
    }
}

// ---------------------------------------------------------------------------
// Flash-style attention: one (b, h, 128 Q rows) per block; 16 Q rows / wave.
// ---------------------------------------------------------------------------
__global__ __launch_bounds__(256)
void attn_kernel(const f16* __restrict__ Qh, const f16* __restrict__ Kh,
                 const f16* __restrict__ Vt, const unsigned char* __restrict__ mask,
                 f16* __restrict__ ctxA)
{
  __shared__ f16 Plds[8][16 * 64];
  const int t    = threadIdx.x;
  const int lane = t & 31, w = t >> 5;
  const int half = lane >> 4, r16 = lane & 15;
  const int b = blockIdx.z, h = blockIdx.y;
  const int q0 = blockIdx.x * 128 + w * 16;
  const size_t bh = (size_t)b * HH + h;

  v16h qa[2];
  {
    const f16* qp = Qh + (bh * SS + q0 + r16) * DKK;
    #pragma unroll
    for (int kf = 0; kf < 2; ++kf) qa[kf] = load_frag_a(qp + kf * 32, half);
  }
  v8f acc[4] = {};
  float mi[8], li[8];
  #pragma unroll
  for (int r = 0; r < 8; ++r) { mi[r] = -1e30f; li[r] = 0.f; }

  for (int j0 = 0; j0 < SS; j0 += 64) {
    v8f s[4];
    #pragma unroll
    for (int nf = 0; nf < 4; ++nf) {
      const f16* kp = Kh + (bh * SS + j0 + nf * 16 + r16) * DKK;
      v8f sv = {};
      sv = wmma16(qa[0], load_frag_b(kp, half), sv);
      sv = wmma16(qa[1], load_frag_b(kp + 32, half), sv);
      s[nf] = sv;
    }
    // scale + mask (C-frag layout: lane holds col r16, rows half*8+r)
    const unsigned char* mrow = mask + ((size_t)b * SS + q0 + half * 8) * SS + j0;
    #pragma unroll
    for (int nf = 0; nf < 4; ++nf)
      #pragma unroll
      for (int r = 0; r < 8; ++r) {
        float v = s[nf][r] * 0.125f;
        if (mrow[(size_t)r * SS + nf * 16 + r16]) v = -1e9f;
        s[nf][r] = v;
      }
    // online softmax; row reductions across the 16-lane half sharing these rows
    float fac[8], rs[8];
    #pragma unroll
    for (int r = 0; r < 8; ++r) {
      float cm = fmaxf(fmaxf(s[0][r], s[1][r]), fmaxf(s[2][r], s[3][r]));
      #pragma unroll
      for (int msk = 1; msk < 16; msk <<= 1)
        cm = fmaxf(cm, __shfl_xor(cm, msk, 32));
      const float nm = fmaxf(mi[r], cm);
      fac[r] = __expf(mi[r] - nm);
      mi[r]  = nm;
      rs[r]  = 0.f;
    }
    #pragma unroll
    for (int nf = 0; nf < 4; ++nf)
      #pragma unroll
      for (int r = 0; r < 8; ++r) {
        const float p = __expf(s[nf][r] - mi[r]);
        s[nf][r] = p;
        rs[r] += p;
      }
    #pragma unroll
    for (int r = 0; r < 8; ++r) {
      #pragma unroll
      for (int msk = 1; msk < 16; msk <<= 1)
        rs[r] += __shfl_xor(rs[r], msk, 32);
      li[r] = li[r] * fac[r] + rs[r];
    }
    #pragma unroll
    for (int dv = 0; dv < 4; ++dv)
      #pragma unroll
      for (int r = 0; r < 8; ++r)
        acc[dv][r] *= fac[r];
    // transpose P (C-layout -> A-layout) through this wave's LDS slab
    f16* pl = &Plds[w][0];
    #pragma unroll
    for (int nf = 0; nf < 4; ++nf)
      #pragma unroll
      for (int r = 0; r < 8; ++r)
        pl[(half * 8 + r) * 64 + nf * 16 + r16] = (f16)s[nf][r];
    v16h pa[2];
    #pragma unroll
    for (int kf = 0; kf < 2; ++kf) pa[kf] = load_frag_a(pl + r16 * 64 + kf * 32, half);
    // ctx += P x V   (V pre-transposed to [dv][s] so B frags are contiguous)
    #pragma unroll
    for (int dv = 0; dv < 4; ++dv) {
      const f16* vp = Vt + (bh * DKK + dv * 16 + r16) * SS + j0;
      acc[dv] = wmma16(pa[0], load_frag_b(vp, half), acc[dv]);
      acc[dv] = wmma16(pa[1], load_frag_b(vp + 32, half), acc[dv]);
    }
  }
  // normalized context, written f16 in [b, s, h*64+dv] (A-panel for Wo GEMM)
  #pragma unroll
  for (int dv = 0; dv < 4; ++dv)
    #pragma unroll
    for (int r = 0; r < 8; ++r) {
      const int qr  = q0 + half * 8 + r;
      const int col = h * DKK + dv * 16 + r16;
      ctxA[((size_t)b * SS + qr) * DD + col] = (f16)(acc[dv][r] / li[r]);
    }
}

// ---------------------------------------------------------------------------
// Fused residual add + LayerNorm (f32), optional f16 copy for next GEMM.
// ---------------------------------------------------------------------------
__global__ __launch_bounds__(256)
void residual_ln_kernel(const float* __restrict__ a, const float* __restrict__ res,
                        const float* __restrict__ g, const float* __restrict__ beta,
                        float* __restrict__ outf, f16* __restrict__ outh)
{
  __shared__ float red[256];
  const int row = blockIdx.x, t = threadIdx.x;
  const size_t base = (size_t)row * DD;
  float v[4]; float s = 0.f, s2 = 0.f;
  #pragma unroll
  for (int k = 0; k < 4; ++k) {
    const int i = t + k * 256;
    v[k] = a[base + i] + res[base + i];
    s += v[k]; s2 += v[k] * v[k];
  }
  red[t] = s; __syncthreads();
  for (int o = 128; o > 0; o >>= 1) { if (t < o) red[t] += red[t + o]; __syncthreads(); }
  const float mu = red[0] / DD; __syncthreads();
  red[t] = s2; __syncthreads();
  for (int o = 128; o > 0; o >>= 1) { if (t < o) red[t] += red[t + o]; __syncthreads(); }
  const float var = red[0] / DD - mu * mu;
  const float inv = rsqrtf(var + 1e-6f);
  #pragma unroll
  for (int k = 0; k < 4; ++k) {
    const int i = t + k * 256;
    const float y = (v[k] - mu) * inv * g[i] + beta[i];
    outf[base + i] = y;
    if (outh) outh[base + i] = (f16)y;
  }
}

// ---------------------------------------------------------------------------
// Packing helpers
// ---------------------------------------------------------------------------
__global__ void transpose_convert_kernel(const float* __restrict__ src, f16* __restrict__ dst,
                                         int K, int N)  // src[K][N] -> dst[N][K] f16
{
  const size_t idx = (size_t)blockIdx.x * 256 + threadIdx.x;
  const int n = (int)(idx / K), k = (int)(idx % K);
  dst[idx] = (f16)src[(size_t)k * N + n];
}

__global__ void convert_kernel(const float* __restrict__ src, f16* __restrict__ dst)
{
  const size_t i = (size_t)blockIdx.x * 256 + threadIdx.x;
  dst[i] = (f16)src[i];
}

// q/k/v f16 [b,s,h,64] -> Qh/Kh [b,h,s,64], Vt [b,h,64,s]
__global__ void pack_qkv_kernel(const f16* __restrict__ q, const f16* __restrict__ k,
                                const f16* __restrict__ v,
                                f16* __restrict__ Qh, f16* __restrict__ Kh, f16* __restrict__ Vt)
{
  const size_t idx = (size_t)blockIdx.x * 256 + threadIdx.x;
  const int d  = (int)(idx & 63);
  const int hh = (int)((idx >> 6) & (HH - 1));
  const int s  = (int)((idx >> 10) & (SS - 1));
  const int b  = (int)(idx >> 21);
  const size_t bh = (size_t)b * HH + hh;
  Qh[(bh * SS + s) * 64 + d] = q[idx];
  Kh[(bh * SS + s) * 64 + d] = k[idx];
  Vt[(bh * 64 + d) * (size_t)SS + s] = v[idx];
}

// ---------------------------------------------------------------------------
extern "C" void kernel_launch(void* const* d_in, const int* in_sizes, int n_in,
                              void* d_out, int out_size, void* d_ws, size_t ws_size,
                              hipStream_t stream)
{
  (void)in_sizes; (void)n_in; (void)out_size; (void)ws_size;
  const float* src  = (const float*)d_in[0];
  const unsigned char* mask = (const unsigned char*)d_in[1];
  const float* Wq = (const float*)d_in[2];  const float* bq = (const float*)d_in[3];
  const float* Wk = (const float*)d_in[4];  const float* bk = (const float*)d_in[5];
  const float* Wv = (const float*)d_in[6];  const float* bv = (const float*)d_in[7];
  const float* Wo = (const float*)d_in[8];  const float* bo = (const float*)d_in[9];
  const float* ln1_g = (const float*)d_in[10]; const float* ln1_b = (const float*)d_in[11];
  const float* W1 = (const float*)d_in[12]; const float* b1 = (const float*)d_in[13];
  const float* W2 = (const float*)d_in[14]; const float* b2 = (const float*)d_in[15];
  const float* ln2_g = (const float*)d_in[16]; const float* ln2_b = (const float*)d_in[17];

  char* ws = (char*)d_ws;
  const size_t MBy = 1024 * 1024;
  f16* WqT  = (f16*)(ws +   0 * MBy);
  f16* WkT  = (f16*)(ws +   2 * MBy);
  f16* WvT  = (f16*)(ws +   4 * MBy);
  f16* WoT  = (f16*)(ws +   6 * MBy);
  f16* W1T  = (f16*)(ws +   8 * MBy);
  f16* W2T  = (f16*)(ws +  16 * MBy);
  f16* srcA = (f16*)(ws +  24 * MBy);
  f16* qf   = (f16*)(ws +  40 * MBy);
  f16* kf   = (f16*)(ws +  56 * MBy);
  f16* vf   = (f16*)(ws +  72 * MBy);
  f16* Qh   = (f16*)(ws +  88 * MBy);
  f16* Kh   = (f16*)(ws + 104 * MBy);
  f16* Vt   = (f16*)(ws + 120 * MBy);
  f16* ctxA = qf;                          // qf dead after pack_qkv
  float* att = (float*)(ws + 136 * MBy);
  float* x   = (float*)(ws + 168 * MBy);   // live until final LN
  f16* xA  = kf;                           // kf dead after pack_qkv
  f16* hA  = (f16*)(ws + 88 * MBy);        // Qh/Kh/Vt + att dead by GEMM3

  const int M = BB * SS;  // 8192

  // pack weights (transposed f16) and src panel
  transpose_convert_kernel<<<(DD * DD) / 256, 256, 0, stream>>>(Wq, WqT, DD, DD);
  transpose_convert_kernel<<<(DD * DD) / 256, 256, 0, stream>>>(Wk, WkT, DD, DD);
  transpose_convert_kernel<<<(DD * DD) / 256, 256, 0, stream>>>(Wv, WvT, DD, DD);
  transpose_convert_kernel<<<(DD * DD) / 256, 256, 0, stream>>>(Wo, WoT, DD, DD);
  transpose_convert_kernel<<<(DD * FFD) / 256, 256, 0, stream>>>(W1, W1T, DD, FFD);
  transpose_convert_kernel<<<(FFD * DD) / 256, 256, 0, stream>>>(W2, W2T, FFD, DD);
  convert_kernel<<<(M * DD) / 256, 256, 0, stream>>>(src, srcA);

  // QKV projections (f16 out for attention)
  dim3 g1(DD / 128, M / 128);
  gemm_kernel<true,  false><<<g1, 256, 0, stream>>>(srcA, WqT, bq, qf, M, DD, DD);
  gemm_kernel<true,  false><<<g1, 256, 0, stream>>>(srcA, WkT, bk, kf, M, DD, DD);
  gemm_kernel<true,  false><<<g1, 256, 0, stream>>>(srcA, WvT, bv, vf, M, DD, DD);
  pack_qkv_kernel<<<(M * DD) / 256, 256, 0, stream>>>(qf, kf, vf, Qh, Kh, Vt);

  // attention -> ctx f16 A-panel
  attn_kernel<<<dim3(SS / 128, HH, BB), 256, 0, stream>>>(Qh, Kh, Vt, mask, ctxA);

  // output projection, residual + LN1 (f32 x + f16 panel)
  gemm_kernel<false, false><<<g1, 256, 0, stream>>>(ctxA, WoT, bo, att, M, DD, DD);
  residual_ln_kernel<<<M, 256, 0, stream>>>(att, src, ln1_g, ln1_b, x, xA);

  // FFN: relu(x W1 + b1) f16, then W2 into d_out, residual + LN2 in place
  gemm_kernel<true,  true ><<<dim3(FFD / 128, M / 128), 256, 0, stream>>>(xA, W1T, b1, hA, M, FFD, DD);
  gemm_kernel<false, false><<<g1, 256, 0, stream>>>(hA, W2T, b2, (float*)d_out, M, DD, FFD);
  residual_ln_kernel<<<M, 256, 0, stream>>>((const float*)d_out, x, ln2_g, ln2_b, (float*)d_out, nullptr);
}
